// TrjAttention_61083024883912
// MI455X (gfx1250) — matrix operand order repaired and checked by reading, not compiled, not run on hardware
//
#include <hip/hip_runtime.h>

// ---------------- types ----------------
typedef __bf16 bf16_t;
typedef bf16_t v16bf __attribute__((ext_vector_type(16)));
typedef float  v8f   __attribute__((ext_vector_type(8)));

__device__ __forceinline__ bf16_t f2bf(float f) {
    unsigned u = __builtin_bit_cast(unsigned, f);
    unsigned r = u + 0x7fffu + ((u >> 16) & 1u);
    unsigned short h = (unsigned short)(r >> 16);
    return __builtin_bit_cast(bf16_t, h);
}
__device__ __forceinline__ float bf2f(bf16_t b) {
    unsigned short h = __builtin_bit_cast(unsigned short, b);
    unsigned u = ((unsigned)h) << 16;
    return __builtin_bit_cast(float, u);
}

// ---- CDNA5 async DMA: global -> LDS (ASYNCcnt-tracked, 16B per lane) ----
__device__ __forceinline__ void async_ldst_b128(unsigned lds_off, const void* gaddr) {
    asm volatile("global_load_async_to_lds_b128 %0, %1, off"
                 :: "v"(lds_off), "v"(gaddr) : "memory");
}
__device__ __forceinline__ void wait_async0() {
    asm volatile("s_wait_asynccnt 0" ::: "memory");
}

// One 16x16 f32 tile accumulated over K (multiple of 32) with bf16 WMMA.
__device__ __forceinline__ v8f tile_mm(const bf16_t* Arow, const bf16_t* Bcol, int K) {
    v8f acc = {0.f,0.f,0.f,0.f,0.f,0.f,0.f,0.f};
    for (int k = 0; k < K; k += 32) {
        v16bf a, b;
#pragma unroll
        for (int i = 0; i < 8; ++i) { a[i] = Arow[k + i]; a[i + 8] = Arow[k + 16 + i]; }
#pragma unroll
        for (int i = 0; i < 8; ++i) { b[i] = Bcol[k - 8 * ((threadIdx.x >> 4) & 1) + i]; b[i + 8] = Bcol[k + 16 - 8 * ((threadIdx.x >> 4) & 1) + i]; }
        acc = __builtin_amdgcn_wmma_f32_16x16x32_bf16(false, a, false, b, (short)0, acc,
                                                      false, false);
    }
    return acc;
}

// ---------------- constants ----------------
#define C_H 8
#define C_DH 64
#define C_NF 16
#define C_P 196
#define C_PP 224
#define C_L 128
#define C_DIM 512
#define C_N 3136
#define C_B 2
#define C_BH 16
#define C_NTOK 3137
#define C_MTOT 6274
#define C_SCALE 0.125f

// ---------------- prep kernels ----------------
__global__ void k_transpose(const float* in, int ldin, int K, int Ncols, bf16_t* out) {
    long i = (long)blockIdx.x * blockDim.x + threadIdx.x;
    long total = (long)K * Ncols;
    if (i >= total) return;
    int r = (int)(i / Ncols), c = (int)(i % Ncols);
    out[(long)c * K + r] = f2bf(in[(long)r * ldin + c]);
}

__global__ void k_f2bf(const float* in, bf16_t* out, long n) {
    long i = (long)blockIdx.x * blockDim.x + threadIdx.x;
    if (i < n) out[i] = f2bf(in[i]);
}

__global__ void k_zero(bf16_t* p, long n) {
    long i = (long)blockIdx.x * blockDim.x + threadIdx.x;
    if (i < n) p[i] = f2bf(0.f);
}

// ---------------- 1) QKV GEMM: (6274x512)@(512x1536) ----------------
__global__ void k_qkv(const bf16_t* __restrict__ X, const bf16_t* __restrict__ Wt,
                      bf16_t* __restrict__ Q, bf16_t* __restrict__ Kb,
                      bf16_t* __restrict__ V, bf16_t* __restrict__ VT) {
    const int NT = 1536 / 16;                       // 96
    const int MT = (C_MTOT + 15) / 16;              // 393
    long tile = (long)blockIdx.x * 8 + (threadIdx.x >> 5);
    if (tile >= (long)MT * NT) return;
    int mt = (int)(tile / NT), nt = (int)(tile % NT);
    int lane = threadIdx.x & 31;
    int r = mt * 16 + (lane & 15);
    int rc = r < C_MTOT ? r : C_MTOT - 1;
    const bf16_t* Arow = X + (long)rc * C_DIM + ((lane >> 4) << 3);
    const bf16_t* Bcol = Wt + ((long)(nt * 16 + (lane & 15))) * C_DIM + ((lane >> 4) << 4);
    v8f acc = tile_mm(Arow, Bcol, C_DIM);
    int c = nt * 16 + (lane & 15);
    int t = c >> 9, h = (c >> 6) & 7, d = c & 63;
    int rb = mt * 16 + ((lane >> 4) << 3);
#pragma unroll
    for (int i = 0; i < 8; ++i) {
        int m = rb + i;
        if (m >= C_MTOT) continue;
        int bb = m / C_NTOK, n = m % C_NTOK;
        long bh = (long)bb * C_H + h;
        bf16_t val = f2bf(acc[i]);
        bf16_t* dst = (t == 0) ? Q : (t == 1) ? Kb : V;
        dst[(bh * C_NTOK + n) * C_DH + d] = val;
        if (t == 2 && n >= 1) {
            int pn = n - 1, f = pn / C_P, p = pn % C_P;
            VT[(bh * C_DH + d) * (C_NF * C_PP) + f * C_PP + p] = val;
        }
    }
}

// ---------------- 2) cls-token attention ----------------
__global__ void k_cls(const bf16_t* __restrict__ Q, const bf16_t* __restrict__ Kb,
                      const bf16_t* __restrict__ V, bf16_t* __restrict__ PRE) {
    int bh = blockIdx.x, bb = bh >> 3, h = bh & 7;
    __shared__ float qrow[C_DH];
    __shared__ float logits[C_NTOK];
    __shared__ float red[256];
    const bf16_t* qb = Q + (long)bh * C_NTOK * C_DH;
    if (threadIdx.x < C_DH) qrow[threadIdx.x] = bf2f(qb[threadIdx.x]);
    __syncthreads();
    for (int j = threadIdx.x; j < C_NTOK; j += 256) {
        const bf16_t* kr = Kb + ((long)bh * C_NTOK + j) * C_DH;
        float s = 0.f;
        for (int d = 0; d < C_DH; ++d) s += qrow[d] * bf2f(kr[d]);
        logits[j] = s * C_SCALE;
    }
    __syncthreads();
    float m = -1e30f;
    for (int j = threadIdx.x; j < C_NTOK; j += 256) m = fmaxf(m, logits[j]);
    red[threadIdx.x] = m; __syncthreads();
    for (int s = 128; s > 0; s >>= 1) {
        if (threadIdx.x < s) red[threadIdx.x] = fmaxf(red[threadIdx.x], red[threadIdx.x + s]);
        __syncthreads();
    }
    m = red[0]; __syncthreads();
    float sum = 0.f;
    for (int j = threadIdx.x; j < C_NTOK; j += 256) {
        float e = __expf(logits[j] - m); logits[j] = e; sum += e;
    }
    red[threadIdx.x] = sum; __syncthreads();
    for (int s = 128; s > 0; s >>= 1) {
        if (threadIdx.x < s) red[threadIdx.x] += red[threadIdx.x + s];
        __syncthreads();
    }
    float inv = 1.f / red[0];
    if (threadIdx.x < C_DH) {
        int d = threadIdx.x; float a = 0.f;
        for (int j = 0; j < C_NTOK; ++j)
            a += logits[j] * bf2f(V[((long)bh * C_NTOK + j) * C_DH + d]);
        PRE[((long)bb * C_NTOK) * C_DIM + h * C_DH + d] = f2bf(a * inv);
    }
}

// ---------------- 3) greedy orthogonal landmark selection ----------------
__global__ void k_landmarks(const bf16_t* __restrict__ Q, bf16_t* __restrict__ LM) {
    int bh = blockIdx.x;
    const bf16_t* qb = Q + ((long)bh * C_NTOK + 1) * C_DH;
    __shared__ float maxsim[C_N];
    __shared__ float invn[C_N];
    __shared__ float cur[C_DH];
    __shared__ float redv[256];
    __shared__ int   redi[256];
    __shared__ int   sel_idx;
    for (int n = threadIdx.x; n < C_N; n += 256) {
        const bf16_t* r = qb + (long)n * C_DH;
        float s = 0.f;
        for (int d = 0; d < C_DH; ++d) { float x = bf2f(r[d]); s += x * x; }
        invn[n] = 1.f / fmaxf(sqrtf(s), 1e-12f);
        maxsim[n] = 0.f;
    }
    if (threadIdx.x == 0) sel_idx = 0;
    __syncthreads();
    int count = 0;
    while (true) {
        int idx = sel_idx;
        if (threadIdx.x < C_DH) {
            bf16_t qv = qb[(long)idx * C_DH + threadIdx.x];
            LM[((long)bh * C_L + count) * C_DH + threadIdx.x] = qv;
            cur[threadIdx.x] = bf2f(qv) * invn[idx];
        }
        if (threadIdx.x == 0) maxsim[idx] = 10.f;
        ++count;
        if (count == C_L) break;
        __syncthreads();
        float best = 1e30f; int bidx = 0;
        for (int n = threadIdx.x; n < C_N; n += 256) {
            float ms = maxsim[n];
            if (ms < 9.f) {
                const bf16_t* r = qb + (long)n * C_DH;
                float s = 0.f;
                for (int d = 0; d < C_DH; ++d) s += bf2f(r[d]) * cur[d];
                ms = fmaxf(ms, fabsf(s) * invn[n]);
                maxsim[n] = ms;
            }
            if (ms < best) { best = ms; bidx = n; }
        }
        redv[threadIdx.x] = best; redi[threadIdx.x] = bidx;
        __syncthreads();
        for (int s = 128; s > 0; s >>= 1) {
            if (threadIdx.x < s && redv[threadIdx.x + s] < redv[threadIdx.x]) {
                redv[threadIdx.x] = redv[threadIdx.x + s];
                redi[threadIdx.x] = redi[threadIdx.x + s];
            }
            __syncthreads();
        }
        if (threadIdx.x == 0) sel_idx = redi[0];
        __syncthreads();
    }
}

// ---------------- 4) kernel_1 = softmax(q_patch @ lm^T), fused ----------------
__global__ void k_kernel1(const bf16_t* __restrict__ Q, const bf16_t* __restrict__ LM,
                          bf16_t* __restrict__ K1) {
    int blk = blockIdx.x;
    int bh = blk / (C_N / 16), mt = blk % (C_N / 16);
    int lane = threadIdx.x & 31, wave = threadIdx.x >> 5;
    const bf16_t* Arow = Q + ((long)bh * C_NTOK + 1 + mt * 16 + (lane & 15)) * C_DH
                         + ((lane >> 4) << 3);
    const bf16_t* Bcol = LM + ((long)bh * C_L + wave * 16 + (lane & 15)) * C_DH
                         + ((lane >> 4) << 4);
    v8f acc = tile_mm(Arow, Bcol, C_DH);
    __shared__ float tileS[16][C_L];
    int cc = wave * 16 + (lane & 15);
    int rb = (lane >> 4) << 3;
#pragma unroll
    for (int i = 0; i < 8; ++i) tileS[rb + i][cc] = acc[i];
    __syncthreads();
    if (threadIdx.x < 16) {
        int r = threadIdx.x;
        float mx = -1e30f;
        for (int c = 0; c < C_L; ++c) mx = fmaxf(mx, tileS[r][c]);
        float sm = 0.f;
        for (int c = 0; c < C_L; ++c) { float e = __expf(tileS[r][c] - mx); tileS[r][c] = e; sm += e; }
        float inv = 1.f / sm;
        bf16_t* dst = K1 + ((long)bh * C_N + mt * 16 + r) * C_L;
        for (int c = 0; c < C_L; ++c) dst[c] = f2bf(tileS[r][c] * inv);
    }
}

// ---------------- 5a) lk = lm @ k_patch^T ----------------
__global__ void k_lk(const bf16_t* __restrict__ LM, const bf16_t* __restrict__ Kb,
                     float* __restrict__ LK) {
    const int NT = C_N / 16;
    const int TPB = (C_L / 16) * NT;
    long tile = (long)blockIdx.x * 8 + (threadIdx.x >> 5);
    int bh = (int)(tile / TPB); int rem = (int)(tile % TPB);
    int mt = rem / NT, nt = rem % NT;
    int lane = threadIdx.x & 31;
    const bf16_t* Arow = LM + ((long)bh * C_L + mt * 16 + (lane & 15)) * C_DH
                         + ((lane >> 4) << 3);
    const bf16_t* Bcol = Kb + ((long)bh * C_NTOK + 1 + nt * 16 + (lane & 15)) * C_DH
                         + ((lane >> 4) << 4);
    v8f acc = tile_mm(Arow, Bcol, C_DH);
    int c = nt * 16 + (lane & 15);
    int rb = mt * 16 + ((lane >> 4) << 3);
#pragma unroll
    for (int i = 0; i < 8; ++i)
        LK[((long)bh * C_L + rb + i) * C_N + c] = acc[i];
}

// ---------------- 5b) softmax over P=196 -> padded bf16 kernel_2 ----------------
__global__ void k_kernel2(const float* __restrict__ LK, bf16_t* __restrict__ K2P) {
    int row = blockIdx.x * 8 + (threadIdx.x >> 5);
    int lane = threadIdx.x & 31;
    int bh = row >> 11, rem = row & 2047, l = rem >> 4, f = rem & 15;
    const float* src = LK + ((long)bh * C_L + l) * C_N + f * C_P;
    float v[7]; float mx = -1e30f;
#pragma unroll
    for (int i = 0; i < 7; ++i) {
        int p = i * 32 + lane;
        v[i] = (p < C_P) ? src[p] : -1e30f;
        mx = fmaxf(mx, v[i]);
    }
    for (int m = 16; m > 0; m >>= 1) mx = fmaxf(mx, __shfl_xor(mx, m, 32));
    float sm = 0.f;
#pragma unroll
    for (int i = 0; i < 7; ++i) {
        float e = (v[i] > -1e29f) ? __expf(v[i] - mx) : 0.f;
        v[i] = e; sm += e;
    }
    for (int m = 16; m > 0; m >>= 1) sm += __shfl_xor(sm, m, 32);
    float inv = 1.f / sm;
    bf16_t* dst = K2P + (((long)bh * C_NF + f) * C_L + l) * C_PP;
#pragma unroll
    for (int i = 0; i < 7; ++i) {
        int p = i * 32 + lane;
        dst[p] = f2bf((p < C_P) ? v[i] * inv : 0.f);
    }
}

// ---------------- 5c) xl = kernel_2 @ v  -> transposed ----------------
__global__ void k_xl(const bf16_t* __restrict__ K2P, const bf16_t* __restrict__ VT,
                     bf16_t* __restrict__ XLT) {
    long tile = (long)blockIdx.x * 8 + (threadIdx.x >> 5);
    int bhf = (int)(tile / 32); int rem = (int)(tile % 32);
    int mt = rem / 4, nt = rem % 4;
    int bh = bhf / C_NF, f = bhf % C_NF;
    int lane = threadIdx.x & 31;
    const bf16_t* Arow = K2P + (((long)bh * C_NF + f) * C_L + mt * 16 + (lane & 15)) * C_PP
                         + ((lane >> 4) << 3);
    const bf16_t* Bcol = VT + ((long)bh * C_DH + nt * 16 + (lane & 15)) * (C_NF * C_PP)
                         + f * C_PP + ((lane >> 4) << 4);
    v8f acc = tile_mm(Arow, Bcol, C_PP);
    int d = nt * 16 + (lane & 15);
    int rb = mt * 16 + ((lane >> 4) << 3);
    bf16_t* dst = XLT + ((long)bh * 1024 + f * C_DH + d) * C_L + rb;
#pragma unroll
    for (int i = 0; i < 8; ++i) dst[i] = f2bf(acc[i]);
}

// ---------------- 6) xo = kernel_1 @ xl ----------------
__global__ void k_xo(const bf16_t* __restrict__ K1, const bf16_t* __restrict__ XLT,
                     bf16_t* __restrict__ XO) {
    const int TPB = (C_N / 16) * (1024 / 16);
    long tile = (long)blockIdx.x * 8 + (threadIdx.x >> 5);
    int bh = (int)(tile / TPB); int rem = (int)(tile % TPB);
    int mt = rem / 64, nt = rem % 64;
    int lane = threadIdx.x & 31;
    const bf16_t* Arow = K1 + ((long)bh * C_N + mt * 16 + (lane & 15)) * C_L
                         + ((lane >> 4) << 3);
    const bf16_t* Bcol = XLT + ((long)bh * 1024 + nt * 16 + (lane & 15)) * C_L
                         + ((lane >> 4) << 4);
    v8f acc = tile_mm(Arow, Bcol, C_L);
    int c = nt * 16 + (lane & 15);
    int f = c >> 6, d = c & 63;
    int bb = bh >> 3, h = bh & 7;
    int rb = mt * 16 + ((lane >> 4) << 3);
#pragma unroll
    for (int i = 0; i < 8; ++i)
        XO[(((long)bb * C_N + rb + i) * C_NF + f) * C_DIM + h * C_DH + d] = f2bf(acc[i]);
}

// ---------------- 7) q2 = x_diag @ W_q * SCALE ----------------
__global__ void k_q2(const bf16_t* __restrict__ XO, const bf16_t* __restrict__ Wt,
                     bf16_t* __restrict__ Q2) {
    const int NT = 32;
    long tile = (long)blockIdx.x * 8 + (threadIdx.x >> 5);
    int mt = (int)(tile / NT), nt = (int)(tile % NT);
    int lane = threadIdx.x & 31;
    int r = mt * 16 + (lane & 15);
    int bb = r / C_N, n = r % C_N, fr = n / C_P;
    const bf16_t* Arow = XO + (((long)bb * C_N + n) * C_NF + fr) * C_DIM + ((lane >> 4) << 3);
    const bf16_t* Bcol = Wt + ((long)(nt * 16 + (lane & 15))) * C_DIM + ((lane >> 4) << 4);
    v8f acc = tile_mm(Arow, Bcol, C_DIM);
    int c = nt * 16 + (lane & 15);
    int rb = mt * 16 + ((lane >> 4) << 3);
#pragma unroll
    for (int i = 0; i < 8; ++i) {
        int rr = rb + i; int b2 = rr / C_N, n2 = rr % C_N;
        Q2[((long)b2 * C_N + n2) * C_DIM + c] = f2bf(acc[i] * C_SCALE);
    }
}

// ---------------- 8) k2 = xo @ W_kv[:, :512] : LDS-staged, async DMA ---------------
// Block tile 128x64, 8 waves, each wave a 16x64 strip (4 accumulators, A-frag reuse).
// K=512 in 8 stages of 64, double-buffered via global_load_async_to_lds_b128.
__global__ void __launch_bounds__(256) k_k2s(const bf16_t* __restrict__ XO,
                                             const bf16_t* __restrict__ Wt,
                                             bf16_t* __restrict__ K2B) {
    __shared__ bf16_t sA[2][128 * 64];
    __shared__ bf16_t sB[2][64 * 64];
    int mblk = blockIdx.x >> 3, nblk = blockIdx.x & 7;   // 784 x 8 blocks
    long mBase = (long)mblk * 128;
    int nBase = nblk * 64;
    int t = threadIdx.x;
    int lane = t & 31, wave = t >> 5;
    unsigned sAoff = (unsigned)(unsigned long long)(&sA[0][0]);
    unsigned sBoff = (unsigned)(unsigned long long)(&sB[0][0]);

    const bf16_t* Abase = XO + mBase * C_DIM;
    const bf16_t* Bbase = Wt + (long)nBase * C_DIM;

    // issue one 64-wide K stage into buffer `buf`
    auto issue_stage = [&](int s, int buf) {
        unsigned aoff = sAoff + (unsigned)buf * (128 * 64 * 2);
#pragma unroll
        for (int j = 0; j < 4; ++j) {
            int chunk = t + 256 * j;                 // 1024 chunks of 8 bf16
            int row = chunk >> 3, kc = chunk & 7;
            async_ldst_b128(aoff + (unsigned)(row * 64 + kc * 8) * 2,
                            Abase + (long)row * C_DIM + s * 64 + kc * 8);
        }
        unsigned boff = sBoff + (unsigned)buf * (64 * 64 * 2);
#pragma unroll
        for (int j = 0; j < 2; ++j) {
            int chunk = t + 256 * j;                 // 512 chunks
            int row = chunk >> 3, kc = chunk & 7;
            async_ldst_b128(boff + (unsigned)(row * 64 + kc * 8) * 2,
                            Bbase + (long)row * C_DIM + s * 64 + kc * 8);
        }
    };

    v8f acc[4];
#pragma unroll
    for (int i = 0; i < 4; ++i) acc[i] = (v8f){0.f,0.f,0.f,0.f,0.f,0.f,0.f,0.f};

    issue_stage(0, 0);
    int buf = 0;
    for (int s = 0; s < 8; ++s) {
        wait_async0();          // this wave's stage-s DMA has landed
        __syncthreads();        // all waves landed; prior compute on other buf done
        if (s + 1 < 8) issue_stage(s + 1, buf ^ 1);
        const bf16_t* aB = &sA[buf][(wave * 16 + (lane & 15)) * 64 + ((lane >> 4) << 3)];
#pragma unroll
        for (int ks = 0; ks < 64; ks += 32) {
            v16bf a;
#pragma unroll
            for (int i = 0; i < 8; ++i) { a[i] = aB[ks + i]; a[i + 8] = aB[ks + 16 + i]; }
#pragma unroll
            for (int nsub = 0; nsub < 4; ++nsub) {
                const bf16_t* bB = &sB[buf][(nsub * 16 + (lane & 15)) * 64
                                            + ((lane >> 4) << 4) + ks];
                v16bf bfr;
#pragma unroll
                for (int i = 0; i < 8; ++i) { bfr[i] = bB[i - ((lane >> 4) << 3)]; bfr[i + 8] = bB[16 - ((lane >> 4) << 3) + i]; }
                acc[nsub] = __builtin_amdgcn_wmma_f32_16x16x32_bf16(
                    false, a, false, bfr, (short)0, acc[nsub], false, false);
            }
        }
        buf ^= 1;
    }

    long rbase = mBase + wave * 16 + ((lane >> 4) << 3);
    int c0 = nBase + (lane & 15);
#pragma unroll
    for (int nsub = 0; nsub < 4; ++nsub)
#pragma unroll
        for (int i = 0; i < 8; ++i)
            K2B[(rbase + i) * C_DIM + c0 + nsub * 16] = f2bf(acc[nsub][i]);
}

// ---------------- 9) frame attention over F=16 ----------------
__global__ void k_attn2(const bf16_t* __restrict__ Q2, const bf16_t* __restrict__ K2B,
                        const bf16_t* __restrict__ XO, bf16_t* __restrict__ PRE) {
    int bn = blockIdx.x;
    int bb = bn / C_N, n = bn % C_N;
    int h = threadIdx.x >> 5, lane = threadIdx.x & 31;
    const bf16_t* q2r = Q2 + ((long)bb * C_N + n) * C_DIM + h * C_DH;
    float q0 = bf2f(q2r[lane * 2]), q1 = bf2f(q2r[lane * 2 + 1]);
    float logit[C_NF];
#pragma unroll
    for (int f = 0; f < C_NF; ++f) {
        const bf16_t* kr = K2B + (((long)bb * C_N + n) * C_NF + f) * C_DIM + h * C_DH;
        float s = q0 * bf2f(kr[lane * 2]) + q1 * bf2f(kr[lane * 2 + 1]);
        for (int m = 16; m > 0; m >>= 1) s += __shfl_xor(s, m, 32);
        logit[f] = s;
    }
    float mx = -1e30f;
#pragma unroll
    for (int f = 0; f < C_NF; ++f) mx = fmaxf(mx, logit[f]);
    float sm = 0.f;
#pragma unroll
    for (int f = 0; f < C_NF; ++f) { logit[f] = __expf(logit[f] - mx); sm += logit[f]; }
    float inv = 1.f / sm;
    float o0 = 0.f, o1 = 0.f;
#pragma unroll
    for (int f = 0; f < C_NF; ++f) {
        const bf16_t* xr = XO + (((long)bb * C_N + n) * C_NF + f) * C_DIM + h * C_DH;
        float w = logit[f] * inv;
        o0 += w * bf2f(xr[lane * 2]);
        o1 += w * bf2f(xr[lane * 2 + 1]);
    }
    bf16_t* dst = PRE + ((long)bb * C_NTOK + 1 + n) * C_DIM + h * C_DH + lane * 2;
    dst[0] = f2bf(o0);
    dst[1] = f2bf(o1);
}

// ---------------- 10) out = [cls; attn] @ W_out + b_out ----------------
__global__ void k_final(const bf16_t* __restrict__ PRE, const bf16_t* __restrict__ Wt,
                        const float* __restrict__ bias, float* __restrict__ OUT) {
    const int NT = 32;
    const int MT = (C_MTOT + 15) / 16;
    long tile = (long)blockIdx.x * 8 + (threadIdx.x >> 5);
    if (tile >= (long)MT * NT) return;
    int mt = (int)(tile / NT), nt = (int)(tile % NT);
    int lane = threadIdx.x & 31;
    int r = mt * 16 + (lane & 15);
    int rc = r < C_MTOT ? r : C_MTOT - 1;
    const bf16_t* Arow = PRE + (long)rc * C_DIM + ((lane >> 4) << 3);
    const bf16_t* Bcol = Wt + ((long)(nt * 16 + (lane & 15))) * C_DIM + ((lane >> 4) << 4);
    v8f acc = tile_mm(Arow, Bcol, C_DIM);
    int c = nt * 16 + (lane & 15);
    float bc = bias[c];
    int rb = mt * 16 + ((lane >> 4) << 3);
#pragma unroll
    for (int i = 0; i < 8; ++i) {
        int m = rb + i;
        if (m < C_MTOT) OUT[(long)m * C_DIM + c] = acc[i] + bc;
    }
}

// ---------------- host launch ----------------
extern "C" void kernel_launch(void* const* d_in, const int* in_sizes, int n_in,
                              void* d_out, int out_size, void* d_ws, size_t ws_size,
                              hipStream_t stream) {
    const float* x     = (const float*)d_in[0];
    const float* W_qkv = (const float*)d_in[1];
    const float* W_q   = (const float*)d_in[2];
    const float* W_kv  = (const float*)d_in[3];
    const float* W_out = (const float*)d_in[4];
    const float* b_out = (const float*)d_in[5];
    float* out = (float*)d_out;

    char* ws = (char*)d_ws;
    size_t off = 0;
    auto alloc = [&](size_t bytes) -> void* {
        void* p = ws + off;
        off += (bytes + 255) & ~(size_t)255;
        return p;
    };
    bf16_t* WQKVT = (bf16_t*)alloc((size_t)1536 * 512 * 2);
    bf16_t* WQT   = (bf16_t*)alloc((size_t)512 * 512 * 2);
    bf16_t* WKVKT = (bf16_t*)alloc((size_t)512 * 512 * 2);
    bf16_t* WOUTT = (bf16_t*)alloc((size_t)512 * 512 * 2);
    bf16_t* XBF   = (bf16_t*)alloc((size_t)C_MTOT * C_DIM * 2);
    bf16_t* QBF   = (bf16_t*)alloc((size_t)C_BH * C_NTOK * C_DH * 2);
    bf16_t* KBF   = (bf16_t*)alloc((size_t)C_BH * C_NTOK * C_DH * 2);
    bf16_t* VBF   = (bf16_t*)alloc((size_t)C_BH * C_NTOK * C_DH * 2);
    bf16_t* VT    = (bf16_t*)alloc((size_t)C_BH * C_DH * C_NF * C_PP * 2);
    bf16_t* LM    = (bf16_t*)alloc((size_t)C_BH * C_L * C_DH * 2);
    bf16_t* K1    = (bf16_t*)alloc((size_t)C_BH * C_N * C_L * 2);
    float*  LK    = (float*)alloc((size_t)C_BH * C_L * C_N * 4);
    bf16_t* K2P   = (bf16_t*)alloc((size_t)C_BH * C_NF * C_L * C_PP * 2);
    bf16_t* XLT   = (bf16_t*)alloc((size_t)C_BH * 1024 * C_L * 2);
    bf16_t* XO    = (bf16_t*)alloc((size_t)C_B * C_N * C_NF * C_DIM * 2);
    bf16_t* Q2    = (bf16_t*)alloc((size_t)C_B * C_N * C_DIM * 2);
    bf16_t* K2B   = (bf16_t*)alloc((size_t)C_B * C_N * C_NF * C_DIM * 2);
    bf16_t* PRE   = (bf16_t*)alloc((size_t)C_B * C_NTOK * C_DIM * 2);

    dim3 blk(256);
    k_transpose<<<(512 * 1536 + 255) / 256, blk, 0, stream>>>(W_qkv, 1536, 512, 1536, WQKVT);
    k_transpose<<<(512 * 512 + 255) / 256, blk, 0, stream>>>(W_q, 512, 512, 512, WQT);
    k_transpose<<<(512 * 512 + 255) / 256, blk, 0, stream>>>(W_kv, 1024, 512, 512, WKVKT);
    k_transpose<<<(512 * 512 + 255) / 256, blk, 0, stream>>>(W_out, 512, 512, 512, WOUTT);
    long nx = (long)C_MTOT * C_DIM;
    k_f2bf<<<(int)((nx + 255) / 256), blk, 0, stream>>>(x, XBF, nx);
    long nvt = (long)C_BH * C_DH * C_NF * C_PP;
    k_zero<<<(int)((nvt + 255) / 256), blk, 0, stream>>>(VT, nvt);

    k_qkv<<<(393 * 96 + 7) / 8, blk, 0, stream>>>(XBF, WQKVT, QBF, KBF, VBF, VT);
    k_cls<<<C_BH, blk, 0, stream>>>(QBF, KBF, VBF, PRE);
    k_landmarks<<<C_BH, blk, 0, stream>>>(QBF, LM);
    k_kernel1<<<C_BH * (C_N / 16), blk, 0, stream>>>(QBF, LM, K1);
    k_lk<<<(C_BH * (C_L / 16) * (C_N / 16)) / 8, blk, 0, stream>>>(LM, KBF, LK);
    k_kernel2<<<(C_BH * C_L * C_NF) / 8, blk, 0, stream>>>(LK, K2P);
    k_xl<<<(C_BH * C_NF * 32) / 8, blk, 0, stream>>>(K2P, VT, XLT);
    k_xo<<<(C_BH * (C_N / 16) * 64) / 8, blk, 0, stream>>>(K1, XLT, XO);
    k_q2<<<((C_B * C_N / 16) * 32) / 8, blk, 0, stream>>>(XO, WQT, Q2);
    k_k2s<<<784 * 8, blk, 0, stream>>>(XO, WKVKT, K2B);
    k_attn2<<<C_B * C_N, blk, 0, stream>>>(Q2, K2B, XO, PRE);
    k_final<<<(393 * 32 + 7) / 8, blk, 0, stream>>>(PRE, WOUTT, b_out, out);
}
